// ks_net_block4_16810501996769
// MI455X (gfx1250) — matrix operand order repaired and checked by reading, not compiled, not run on hardware
//
#include <hip/hip_runtime.h>
#include <hip/hip_fp16.h>
#include <math.h>

#define L    16384
#define NCH  64
#define CLEN 256

typedef __attribute__((ext_vector_type(16))) _Float16 v16h;
typedef __attribute__((ext_vector_type(8)))  float    v8f;
typedef __attribute__((ext_vector_type(4)))  unsigned tdm_v4u;
typedef __attribute__((ext_vector_type(8)))  int      tdm_v8i;
typedef __attribute__((ext_vector_type(4)))  int      tdm_v4i;

__device__ __forceinline__ int   imin_(int a, int b){ return a < b ? a : b; }
__device__ __forceinline__ int   imax_(int a, int b){ return a > b ? a : b; }
__device__ __forceinline__ float sigmoidf_(float x){ return 1.f/(1.f+__expf(-x)); }
__device__ __forceinline__ float siluf_(float x){ return x*sigmoidf_(x); }
__device__ __forceinline__ float softplusf_(float x){ return x>20.f?x:log1pf(__expf(x)); }
__device__ __forceinline__ float geluf_(float x){
  const float c=0.7978845608028654f;
  return 0.5f*x*(1.f+tanhf(c*(x+0.044715f*x*x*x)));
}
// ACT: 0 none, 1 gelu, 2 leaky-relu(0.01), 3 abs  (compile-time)
template<int ACT>
__device__ __forceinline__ float actf_(float v){
  if constexpr (ACT == 1) return geluf_(v);
  if constexpr (ACT == 2) return (v > 0.f) ? v : 0.01f*v;
  if constexpr (ACT == 3) return fabsf(v);
  return v;
}

// ---------------------------------------------------------------------------
// Tensor Data Mover: 1-D copy of nelem f32 from global into LDS at lds_off.
// D# per CDNA5 ISA: group0 = {count|flags, lds_addr,
// global_addr_lo, global_addr_hi|type=2}; group1 = sizes/strides.
// Issued by one wave; completion via TENSORcnt.
// ---------------------------------------------------------------------------
__device__ __forceinline__ void tdm_load_1d(unsigned lds_off, const void* gptr, unsigned nelem)
{
  unsigned long long ga = (unsigned long long)(uintptr_t)gptr;
  tdm_v4u g0;
  g0[0] = 1u;                                            // count=1 (valid user D#)
  g0[1] = lds_off;                                       // lds_addr (bytes)
  g0[2] = (unsigned)(ga & 0xffffffffu);                  // global_addr[31:0]
  g0[3] = (unsigned)((ga >> 32) & 0x01ffffffu) | (2u << 30); // addr[56:32] | type=2
  tdm_v8i g1;
  g1[0] = (int)(2u << 16);                               // data_size=4B, mask=0
  g1[1] = (int)((nelem & 0xffffu) << 16);                // tensor_dim0[15:0]
  g1[2] = (int)(((nelem >> 16) & 0xffffu) | (1u << 16)); // tensor_dim0[31:16], tensor_dim1=1
  g1[3] = (int)((nelem & 0xffffu) << 16);                // tile_dim0 = nelem
  g1[4] = 0;                                             // tile_dim1/2 unused
  g1[5] = (int)nelem;                                    // tensor_dim0_stride[31:0]
  g1[6] = 0;
  g1[7] = 0;
  tdm_v4i gz = {0,0,0,0};
#if __clang_major__ >= 23
  tdm_v8i gz8 = {0,0,0,0,0,0,0,0};
  __builtin_amdgcn_tensor_load_to_lds(g0, g1, gz, gz, gz8, 0);
#else
  __builtin_amdgcn_tensor_load_to_lds(g0, g1, gz, gz, 0);
#endif
}

// ---------------------------------------------------------------------------
// GEMM (K compile-time 16/32): Out[N][L] = A[K][L]^T * W[N][K]^T
// 128 threads = 4 waves, each wave one 16x16 tile (4 consecutive M tiles).
// Weights staged to LDS via TDM; B fragments read from LDS (ds_load).
// ---------------------------------------------------------------------------
template<int K, int NMAX>
__global__ __launch_bounds__(128)
void gemm16_wmma(const float* __restrict__ A,
                 const float* __restrict__ Wt, int N,
                 float* __restrict__ Out)
{
  __shared__ float smw[NMAX * K];
  if (threadIdx.x < 32) {
    tdm_load_1d((unsigned)(uintptr_t)smw, Wt, (unsigned)(N * K));
    __builtin_amdgcn_s_wait_tensorcnt((short)0);
  }
  __syncthreads();

  const int lane = threadIdx.x & 31;
  const int wv   = threadIdx.x >> 5;
  const int m0 = (blockIdx.x * 4 + wv) * 16;
  const int n0 = blockIdx.y * 16;
  const int ml = lane & 15, hi = lane >> 4;
  const int m  = m0 + ml;
  const bool nok = (n0 + ml) < N;
  const int  ncl = nok ? (n0 + ml) : (N - 1);

  v8f acc = {};
  #pragma unroll
  for (int k0 = 0; k0 < K; k0 += 32) {
    v16h a, b;
    #pragma unroll
    for (int e = 0; e < 16; ++e) {
      // A layout (16-bit 16x32): lane halves split K
      int ka = k0 + ((e < 8) ? 0 : 16) + hi*8 + (e & 7);
      float av = 0.f;
      if (ka < K) av = A[(size_t)ka*L + m];              // compile-time predicate
      a[e] = (_Float16)av;
      // B layout (32x16): same K split as A (lane halves split K)
      int kb  = k0 + ((e < 8) ? 0 : 16) + hi*8 + (e & 7);
      int kbc = (kb < K) ? kb : 0;                       // branchless clamp
      float raw = smw[ncl*K + kbc];
      b[e] = (_Float16)(((kb < K) && nok) ? raw : 0.f);
    }
    acc = __builtin_amdgcn_wmma_f32_16x16x32_f16(false, a, false, b, (short)0, acc, false, false);
  }
  #pragma unroll
  for (int r = 0; r < 8; ++r) {
    int row = m0 + r + hi*8;
    int col = n0 + ml;
    if (col < N) Out[(size_t)col*L + row] = acc[r];
  }
}

// ---------------------------------------------------------------------------
// 3x3 SAME conv, on-the-fly im2col + WMMA; Cin compile-time, K = Cin*9.
// STAGE: weights ([Cout][Cin*9], uniform per block) staged to LDS via TDM.
// ACT / RES compile-time -> straight-line epilogue.
// ---------------------------------------------------------------------------
template<int Cin, bool STAGE, int COMAX, int ACT, bool RES>
__global__ __launch_bounds__(128)
void conv3x3_wmma(const float* __restrict__ X,
                  const float* __restrict__ Wt, const float* __restrict__ bias,
                  const float* __restrict__ residual,
                  float* __restrict__ Out, int Cout)
{
  constexpr int K = Cin * 9;
  __shared__ float smw[STAGE ? COMAX * K : 1];
  if constexpr (STAGE) {
    if (threadIdx.x < 32) {
      tdm_load_1d((unsigned)(uintptr_t)smw, Wt, (unsigned)(Cout * K));
      __builtin_amdgcn_s_wait_tensorcnt((short)0);
    }
    __syncthreads();
  }

  const int lane = threadIdx.x & 31;
  const int wv   = threadIdx.x >> 5;
  const int m0 = (blockIdx.x * 4 + wv) * 16;
  const int n0 = blockIdx.y * 16;
  const int ml = lane & 15, hi = lane >> 4;
  const int m  = m0 + ml;
  const int py = m >> 7, px = m & 127;
  const bool nok = (n0 + ml) < Cout;
  const int  ncl = nok ? (n0 + ml) : (Cout - 1);

  v8f acc = {};
  for (int k0 = 0; k0 < K; k0 += 32) {
    if constexpr (!STAGE)
      __builtin_prefetch(&Wt[(size_t)ncl*K + k0], 0, 3);
    v16h a, b;
    #pragma unroll
    for (int e = 0; e < 16; ++e) {
      int ka  = k0 + ((e < 8) ? 0 : 16) + hi*8 + (e & 7);
      int kac = (ka < K) ? ka : 0;
      int ci  = kac / 9, t = kac - ci*9;
      int yy  = py + t/3 - 1, xx = px + (t - (t/3)*3) - 1;
      bool inb = (yy >= 0) & (yy < 128) & (xx >= 0) & (xx < 128) & (ka < K);
      int yyc = imin_(imax_(yy, 0), 127), xxc = imin_(imax_(xx, 0), 127);
      float raw = X[(size_t)ci*L + (yyc << 7) + xxc];    // always in-range
      a[e] = (_Float16)(inb ? raw : 0.f);

      int kb  = k0 + ((e < 8) ? 0 : 16) + hi*8 + (e & 7);
      int kbc = (kb < K) ? kb : 0;
      float wr;
      if constexpr (STAGE) wr = smw[ncl*K + kbc];
      else                 wr = Wt[(size_t)ncl*K + kbc];
      b[e] = (_Float16)(((kb < K) && nok) ? wr : 0.f);
    }
    acc = __builtin_amdgcn_wmma_f32_16x16x32_f16(false, a, false, b, (short)0, acc, false, false);
  }
  #pragma unroll
  for (int r = 0; r < 8; ++r) {
    int row = m0 + r + hi*8;
    int col = n0 + ml;
    if (col < Cout) {
      float v = acc[r] + bias[col];
      if constexpr (RES) v += residual[(size_t)col*L + row];
      Out[(size_t)col*L + row] = actf_<ACT>(v);
    }
  }
}

// ----------------- LayerNorm across channels (channel-major [C][L]) --------
__global__ void layernorm_chan(const float* __restrict__ X, const float* __restrict__ g,
                               const float* __restrict__ b, float* __restrict__ Out, int C)
{
  int l = blockIdx.x*blockDim.x + threadIdx.x;
  if (l >= L) return;
  float v[32];
  float mean = 0.f;
  for (int c = 0; c < C; ++c) { v[c] = X[(size_t)c*L + l]; mean += v[c]; }
  mean /= (float)C;
  float var = 0.f;
  for (int c = 0; c < C; ++c) { float t = v[c]-mean; var += t*t; }
  var /= (float)C;
  float inv = rsqrtf(var + 1e-5f);
  for (int c = 0; c < C; ++c) Out[(size_t)c*L + l] = (v[c]-mean)*inv*g[c] + b[c];
}

// ----------------- Depthwise 3x3 + SiLU (32 channels) ----------------------
__global__ void dwconv3x3_silu(const float* __restrict__ X, const float* __restrict__ Wt,
                               const float* __restrict__ B, float* __restrict__ Out)
{
  int i = blockIdx.x*blockDim.x + threadIdx.x;
  if (i >= 32*L) return;
  int c = i >> 14, l = i & (L-1);
  int py = l >> 7, px = l & 127;
  float s = B[c];
  #pragma unroll
  for (int t = 0; t < 9; ++t) {
    int yy = py + t/3 - 1, xx = px + t%3 - 1;
    bool inb = (yy >= 0) & (yy < 128) & (xx >= 0) & (xx < 128);
    int yyc = imin_(imax_(yy, 0), 127), xxc = imin_(imax_(xx, 0), 127);
    float raw = X[(size_t)c*L + (yyc << 7) + xxc];
    s += Wt[c*9 + t] * (inb ? raw : 0.f);
  }
  Out[i] = siluf_(s);
}

// ----------------- Build the 4 scan directions ------------------------------
__global__ void build_xs(const float* __restrict__ xm, float* __restrict__ xs)
{
  int i = blockIdx.x*blockDim.x + threadIdx.x;
  if (i >= 32*L) return;
  int d = i >> 14, l = i & (L-1);
  float v = xm[i];
  int lt = ((l & 127) << 7) | (l >> 7);         // HxW transpose (128x128)
  xs[(size_t)(0*32 + d)*L + l]         = v;
  xs[(size_t)(1*32 + d)*L + lt]        = v;
  xs[(size_t)(2*32 + d)*L + (L-1-l)]   = v;
  xs[(size_t)(3*32 + d)*L + (L-1-lt)]  = v;
}

// ----------------- Chunked selective scan -----------------------------------
__global__ __launch_bounds__(32)
void scan_passA(const float* __restrict__ xdbl, const float* __restrict__ xs,
                const float* __restrict__ dt_w, const float* __restrict__ dt_b,
                const float* __restrict__ A_log,
                float* __restrict__ chA, float* __restrict__ chH)
{
  int blk = blockIdx.x;            // seq*NCH + chunk
  int seq = blk >> 6, chunk = blk & 63;
  int k = seq >> 5;
  int lane = threadIdx.x & 31, nn = lane & 15;
  float An = -__expf(A_log[seq*16 + nn]);
  float dtw = dt_w[seq], dtb = dt_b[seq];
  const float* dtr = xdbl + ((size_t)k*33)*L;
  const float* Bsp = xdbl + ((size_t)k*33 + 1 + nn)*L;
  const float* xp  = xs + (size_t)seq*L;
  int l0 = chunk*CLEN;
  float prodA = 1.f, h = 0.f;
  for (int i = 0; i < CLEN; ++i) {
    int l = l0 + i;
    float dt = softplusf_(dtr[l]*dtw + dtb);
    float dA = __expf(dt*An);
    h = dA*h + dt*xp[l]*Bsp[l];
    prodA *= dA;
  }
  if (lane < 16) {
    chA[(size_t)blk*16 + nn] = prodA;
    chH[(size_t)blk*16 + nn] = h;
  }
}

__global__ void scan_passB(const float* __restrict__ chA, const float* __restrict__ chH,
                           float* __restrict__ carry)
{
  int seq = blockIdx.x;            // 128
  int nn = threadIdx.x;            // 16 threads
  float c = 0.f;
  for (int ch = 0; ch < NCH; ++ch) {
    size_t idx = ((size_t)seq*NCH + ch)*16 + nn;
    carry[idx] = c;
    c = chA[idx]*c + chH[idx];
  }
}

__global__ __launch_bounds__(32)
void scan_passC(const float* __restrict__ xdbl, const float* __restrict__ xs,
                const float* __restrict__ dt_w, const float* __restrict__ dt_b,
                const float* __restrict__ A_log, const float* __restrict__ Dp,
                const float* __restrict__ carry, float* __restrict__ ysout)
{
  int blk = blockIdx.x;
  int seq = blk >> 6, chunk = blk & 63;
  int k = seq >> 5;
  int lane = threadIdx.x & 31, nn = lane & 15;
  float An = -__expf(A_log[seq*16 + nn]);
  float dtw = dt_w[seq], dtb = dt_b[seq], Dc = Dp[seq];
  const float* dtr = xdbl + ((size_t)k*33)*L;
  const float* Bsp = xdbl + ((size_t)k*33 + 1 + nn)*L;
  const float* Csp = xdbl + ((size_t)k*33 + 17 + nn)*L;
  const float* xp  = xs + (size_t)seq*L;
  float h = carry[(size_t)blk*16 + nn];
  int l0 = chunk*CLEN;
  for (int i = 0; i < CLEN; ++i) {
    int l = l0 + i;
    float xv = xp[l];
    float dt = softplusf_(dtr[l]*dtw + dtb);
    float dA = __expf(dt*An);
    h = dA*h + dt*xv*Bsp[l];
    float s = h * Csp[l];
    s += __shfl_xor(s, 1);
    s += __shfl_xor(s, 2);
    s += __shfl_xor(s, 4);
    s += __shfl_xor(s, 8);
    if (lane == 0) ysout[(size_t)seq*L + l] = s + Dc*xv;
  }
}

// ----------------- Merge 4 directions + out_norm LN + SiLU gate ------------
__global__ void merge_ln_silu(const float* __restrict__ ys, const float* __restrict__ z,
                              const float* __restrict__ g, const float* __restrict__ b,
                              float* __restrict__ Out)
{
  int l = blockIdx.x*blockDim.x + threadIdx.x;
  if (l >= L) return;
  int lt = ((l & 127) << 7) | (l >> 7);
  int lr = L-1-l, lrt = L-1-lt;
  float v[32]; float mean = 0.f;
  #pragma unroll
  for (int d = 0; d < 32; ++d) {
    float s = ys[(size_t)(0*32+d)*L + l] + ys[(size_t)(2*32+d)*L + lr]
            + ys[(size_t)(1*32+d)*L + lt] + ys[(size_t)(3*32+d)*L + lrt];
    v[d] = s; mean += s;
  }
  mean *= (1.f/32.f);
  float var = 0.f;
  #pragma unroll
  for (int d = 0; d < 32; ++d) { float t = v[d]-mean; var += t*t; }
  var *= (1.f/32.f);
  float inv = rsqrtf(var + 1e-5f);
  #pragma unroll
  for (int d = 0; d < 32; ++d) {
    float yv = (v[d]-mean)*inv*g[d] + b[d];
    Out[(size_t)d*L + l] = yv * siluf_(z[(size_t)d*L + l]);
  }
}

// ----------------- out = xin*skip[c] + addend*(addsc?addsc[c]:1) -----------
template<bool SCALED>
__global__ void skip_scale_add(const float* __restrict__ xin, const float* __restrict__ skipc,
                               const float* __restrict__ addend, const float* __restrict__ addsc,
                               float* __restrict__ Out, int C)
{
  int i = blockIdx.x*blockDim.x + threadIdx.x;
  if (i >= C*L) return;
  int c = i / L;
  float a = addend[i];
  if constexpr (SCALED) a *= addsc[c];
  Out[i] = xin[i]*skipc[c] + a;
}

// ----------------- channel mean (global avg pool) ---------------------------
__global__ void chan_mean(const float* __restrict__ X, float* __restrict__ out)
{
  __shared__ float sm[256];
  int c = blockIdx.x;
  float s = 0.f;
  for (int l = threadIdx.x; l < L; l += 256) s += X[(size_t)c*L + l];
  sm[threadIdx.x] = s; __syncthreads();
  for (int o = 128; o > 0; o >>= 1) {
    if ((int)threadIdx.x < o) sm[threadIdx.x] += sm[threadIdx.x + o];
    __syncthreads();
  }
  if (threadIdx.x == 0) out[c] = sm[0] * (1.f/(float)L);
}

// ----------------- tiny channel-attention MLP --------------------------------
__global__ void ca_small(const float* __restrict__ a, const float* __restrict__ w1,
                         const float* __restrict__ b1, const float* __restrict__ w2,
                         const float* __restrict__ b2, float* __restrict__ att)
{
  __shared__ float r;
  if (threadIdx.x == 0) {
    float s = b1[0];
    for (int c = 0; c < 16; ++c) s += w1[c]*a[c];
    r = fmaxf(s, 0.f);
  }
  __syncthreads();
  int c = threadIdx.x;
  if (c < 16) att[c] = sigmoidf_(w2[c]*r + b2[c]);
}

// ===========================================================================
// Host orchestration
// ===========================================================================
struct SS2DP { const float *in_proj_w,*conv_w,*conv_b,*x_proj_w,*dt_w,*dt_b,*A_log,*D,*out_norm_g,*out_norm_b,*out_proj_w; };
struct CABP  { const float *c1_w,*c1_b,*c2_w,*c2_b,*ca1_w,*ca1_b,*ca2_w,*ca2_b; };
struct BlockP{ const float *ln1_g,*ln1_b,*skip1,*ln2_g,*ln2_b,*skip2; SS2DP ss; CABP cab; };

struct Rd { void* const* in; int i; const float* n() { return (const float*)in[i++]; } };

static BlockP read_block(Rd& r) {
  BlockP b;
  b.ln1_g=r.n(); b.ln1_b=r.n(); b.skip1=r.n(); b.ln2_g=r.n(); b.ln2_b=r.n(); b.skip2=r.n();
  b.ss.in_proj_w=r.n(); b.ss.conv_w=r.n(); b.ss.conv_b=r.n(); b.ss.x_proj_w=r.n();
  b.ss.dt_w=r.n(); b.ss.dt_b=r.n(); b.ss.A_log=r.n(); b.ss.D=r.n();
  b.ss.out_norm_g=r.n(); b.ss.out_norm_b=r.n(); b.ss.out_proj_w=r.n();
  b.cab.c1_w=r.n(); b.cab.c1_b=r.n(); b.cab.c2_w=r.n(); b.cab.c2_b=r.n();
  b.cab.ca1_w=r.n(); b.cab.ca1_b=r.n(); b.cab.ca2_w=r.n(); b.cab.ca2_b=r.n();
  return b;
}

struct Ws {
  float *x0,*f,*fw,*t1,*t2,*f2,*xz,*xm,*xs,*xdbl,*ys,*ymrg,*ssout,*cb1,*cb2,*x64;
  float *chA,*chH,*carry,*amean,*att;
};

static void run_vss_block(const BlockP& bp, float* f, const Ws& w, hipStream_t s)
{
  const int TPB = 256;
  const int MB = L/64;  // 4 M-tiles per block
  layernorm_chan<<<dim3((L+TPB-1)/TPB), TPB, 0, s>>>(f, bp.ln1_g, bp.ln1_b, w.t1, 16);
  // in_proj: M=L, K=16, N=64
  gemm16_wmma<16,64><<<dim3(MB, 4), 128, 0, s>>>(w.t1, bp.ss.in_proj_w, 64, w.xz);
  dwconv3x3_silu<<<dim3((32*L+TPB-1)/TPB), TPB, 0, s>>>(w.xz, bp.ss.conv_w, bp.ss.conv_b, w.xm);
  build_xs<<<dim3((32*L+TPB-1)/TPB), TPB, 0, s>>>(w.xm, w.xs);
  // x_proj: M=L, K=32, N=33 (per direction)
  for (int k = 0; k < 4; ++k)
    gemm16_wmma<32,48><<<dim3(MB, 3), 128, 0, s>>>(w.xs + (size_t)k*32*L,
        bp.ss.x_proj_w + (size_t)k*33*32, 33, w.xdbl + (size_t)k*33*L);
  // chunked selective scan
  scan_passA<<<dim3(128*NCH), 32, 0, s>>>(w.xdbl, w.xs, bp.ss.dt_w, bp.ss.dt_b, bp.ss.A_log, w.chA, w.chH);
  scan_passB<<<dim3(128), 16, 0, s>>>(w.chA, w.chH, w.carry);
  scan_passC<<<dim3(128*NCH), 32, 0, s>>>(w.xdbl, w.xs, bp.ss.dt_w, bp.ss.dt_b, bp.ss.A_log, bp.ss.D, w.carry, w.ys);
  merge_ln_silu<<<dim3((L+TPB-1)/TPB), TPB, 0, s>>>(w.ys, w.xz + (size_t)32*L,
      bp.ss.out_norm_g, bp.ss.out_norm_b, w.ymrg);
  // out_proj: M=L, K=32, N=16
  gemm16_wmma<32,16><<<dim3(MB, 1), 128, 0, s>>>(w.ymrg, bp.ss.out_proj_w, 16, w.ssout);
  skip_scale_add<false><<<dim3((16*L+TPB-1)/TPB), TPB, 0, s>>>(f, bp.skip1, w.ssout, nullptr, f, 16);
  // CAB branch
  layernorm_chan<<<dim3((L+TPB-1)/TPB), TPB, 0, s>>>(f, bp.ln2_g, bp.ln2_b, w.t2, 16);
  conv3x3_wmma<16,true,16,1,false><<<dim3(MB, 1), 128, 0, s>>>(w.t2, bp.cab.c1_w, bp.cab.c1_b, nullptr, w.cb1, 5);
  conv3x3_wmma<5,true,16,0,false><<<dim3(MB, 1), 128, 0, s>>>(w.cb1, bp.cab.c2_w, bp.cab.c2_b, nullptr, w.cb2, 16);
  chan_mean<<<dim3(16), 256, 0, s>>>(w.cb2, w.amean);
  ca_small<<<dim3(1), 16, 0, s>>>(w.amean, bp.cab.ca1_w, bp.cab.ca1_b, bp.cab.ca2_w, bp.cab.ca2_b, w.att);
  skip_scale_add<true><<<dim3((16*L+TPB-1)/TPB), TPB, 0, s>>>(f, bp.skip2, w.cb2, w.att, f, 16);
}

extern "C" void kernel_launch(void* const* d_in, const int* in_sizes, int n_in,
                              void* d_out, int out_size, void* d_ws, size_t ws_size,
                              hipStream_t stream)
{
  (void)in_sizes; (void)n_in; (void)out_size; (void)ws_size;
  Rd r{d_in, 0};
  const float* x            = r.n();
  const float* conv_first_w = r.n();
  const float* conv_first_b = r.n();
  const float* pe_g = r.n();
  const float* pe_b = r.n();
  BlockP g0b0 = read_block(r);
  const float* g0_conv_w = r.n(); const float* g0_conv_b = r.n();
  BlockP g1b0 = read_block(r);
  BlockP g1b1 = read_block(r);
  const float* g1_conv_w = r.n(); const float* g1_conv_b = r.n();
  const float* norm_g = r.n(); const float* norm_b = r.n();
  const float* cab_w = r.n();  const float* cab_b = r.n();
  const float* cbu_w = r.n();  const float* cbu_b = r.n();
  const float* conv_last_w = r.n(); const float* conv_last_b = r.n();

  // ---- workspace bump allocator (floats) ----
  float* base = (float*)d_ws;
  size_t off = 0;
  auto alloc = [&](size_t nfl){ float* p = base + off; off += nfl; return p; };
  Ws w;
  w.x0   = alloc((size_t)16*L);
  w.f    = alloc((size_t)16*L);
  w.fw   = alloc((size_t)16*L);
  w.t1   = alloc((size_t)16*L);
  w.t2   = alloc((size_t)16*L);
  w.f2   = alloc((size_t)16*L);
  w.xz   = alloc((size_t)64*L);
  w.xm   = alloc((size_t)32*L);
  w.xs   = alloc((size_t)128*L);
  w.xdbl = alloc((size_t)132*L);
  w.ys   = alloc((size_t)128*L);
  w.ymrg = alloc((size_t)32*L);
  w.ssout= alloc((size_t)16*L);
  w.cb1  = alloc((size_t)16*L);
  w.cb2  = alloc((size_t)16*L);
  w.x64  = alloc((size_t)64*L);
  w.chA  = alloc((size_t)128*NCH*16);
  w.chH  = alloc((size_t)128*NCH*16);
  w.carry= alloc((size_t)128*NCH*16);
  w.amean= alloc(64);
  w.att  = alloc(64);

  const int TPB = 256;
  const int MB = L/64;
  // conv_first: 65 -> 16 (weights 37KB -> LDS staged via TDM)
  conv3x3_wmma<65,true,16,0,false><<<dim3(MB, 1), 128, 0, stream>>>(x, conv_first_w, conv_first_b, nullptr, w.x0, 16);
  layernorm_chan<<<dim3((L+TPB-1)/TPB), TPB, 0, stream>>>(w.x0, pe_g, pe_b, w.f, 16);

  // group 0 (depth 1)
  hipMemcpyAsync(w.fw, w.f, (size_t)16*L*sizeof(float), hipMemcpyDeviceToDevice, stream);
  run_vss_block(g0b0, w.fw, w, stream);
  conv3x3_wmma<16,true,16,0,true><<<dim3(MB, 1), 128, 0, stream>>>(w.fw, g0_conv_w, g0_conv_b, w.f, w.f, 16);

  // group 1 (depth 2)
  hipMemcpyAsync(w.fw, w.f, (size_t)16*L*sizeof(float), hipMemcpyDeviceToDevice, stream);
  run_vss_block(g1b0, w.fw, w, stream);
  run_vss_block(g1b1, w.fw, w, stream);
  conv3x3_wmma<16,true,16,0,true><<<dim3(MB, 1), 128, 0, stream>>>(w.fw, g1_conv_w, g1_conv_b, w.f, w.f, 16);

  // tail
  layernorm_chan<<<dim3((L+TPB-1)/TPB), TPB, 0, stream>>>(w.f, norm_g, norm_b, w.t1, 16);
  conv3x3_wmma<16,true,16,0,true><<<dim3(MB, 1), 128, 0, stream>>>(w.t1, cab_w, cab_b, w.x0, w.f2, 16);
  conv3x3_wmma<16,true,64,2,false><<<dim3(MB, 4), 128, 0, stream>>>(w.f2, cbu_w, cbu_b, nullptr, w.x64, 64);  // leaky relu
  conv3x3_wmma<64,false,1,3,false><<<dim3(MB, 5), 128, 0, stream>>>(w.x64, conv_last_w, conv_last_b, nullptr,
                                                                    (float*)d_out, 65);                        // abs
}
